// gcnmask_27058293965355
// MI455X (gfx1250) — matrix operand order repaired and checked by reading, not compiled, not run on hardware
//
#include <hip/hip_runtime.h>
#include <hip/hip_bf16.h>

// ---------- types ----------
typedef _Float16 v16h __attribute__((ext_vector_type(16)));
typedef _Float16 half8 __attribute__((ext_vector_type(8)));
typedef _Float16 half4 __attribute__((ext_vector_type(4)));
typedef float    v8f  __attribute__((ext_vector_type(8)));

#define FDIM 128
#define TILE_E 16
#define NT 4            // edge tiles per wave in k_mask_agg (B-fragment reuse factor)

// ---------- helpers ----------
// A-fragment loader: 16x32 f16 A tile, per-lane row m (= lane&15) fixed.
// Layout (ISA 7.12.2): lanes 0-15 (khalf=0): VGPR0-3 => K=0..7, VGPR4-7 => K=16..23
//                      lanes 16-31 (khalf=1): VGPR0-3 => K=8..15, VGPR4-7 => K=24..31
__device__ __forceinline__ v16h load_A_frag(const _Float16* rowBase, int kOff) {
    half8 lo = *(const half8*)(rowBase + kOff);
    half8 hi = *(const half8*)(rowBase + kOff + 16);
    v16h a;
#pragma unroll
    for (int i = 0; i < 8; ++i) { a[i] = lo[i]; a[i + 8] = hi[i]; }
    return a;
}

// Fast sigmoid: v_exp_f32 + v_add + v_rcp_f32 (no IEEE div fixup sequence).
__device__ __forceinline__ float fast_sigmoid(float v) {
    return __builtin_amdgcn_rcpf(1.0f + __expf(-v));
}

// Deterministic counter hash; keep with prob 0.4 (dropout p=0.6), integer compare.
__device__ __forceinline__ bool keep_elem(unsigned v) {
    v ^= 42u * 0x9E3779B9u;
    v *= 0x85EBCA6Bu; v ^= v >> 13;
    v *= 0xC2B2AE35u; v ^= v >> 16;
    return v >= 2576980378u;   // floor(0.6 * 2^32)
}

// ---------- K0: init xnew = x, xh = f16(x), out = 0, convert weights to f16 ----------
__global__ void k_init(const float* __restrict__ x,
                       const float* __restrict__ wmask,
                       const float* __restrict__ w0,
                       float* __restrict__ xnew,
                       _Float16* __restrict__ xh,
                       _Float16* __restrict__ wmaskh,
                       _Float16* __restrict__ w0h,
                       float* __restrict__ out,
                       int nNodeF) {
    int i = blockIdx.x * blockDim.x + threadIdx.x;
    if (i < nNodeF) {
        float v = x[i];
        xnew[i] = v;
        xh[i]   = (_Float16)v;
        out[i]  = 0.0f;
    }
    if (i < 2 * FDIM * FDIM) wmaskh[i] = (_Float16)wmask[i];
    if (i < FDIM * FDIM)     w0h[i]    = (_Float16)w0[i];
}

// ---------- K1: per-edge mask GEMM + sigmoid + dropout + m*nei scatter-add ----------
// One wave per group of NT 16-edge tiles. Features gathered as pre-converted f16
// rows (halves L2 traffic, no converts in hot path). B fragments hoisted per j.
__global__ __launch_bounds__(32, 4) void k_mask_agg(
    const _Float16* __restrict__ xh,       // [N,128] f16 feature rows
    const long long* __restrict__ src,
    const long long* __restrict__ dst,
    const _Float16* __restrict__ wmaskh,   // [256,128] f16 row-major
    float* __restrict__ xnew,              // pre-initialized to x; accumulates agg
    int nEdges, int nGroups) {
    __shared__ int srcL[NT * TILE_E];
    __shared__ int dstL[NT * TILE_E];
    __shared__ __align__(16) _Float16 Abuf[NT * TILE_E][264];  // [64][256] cen|nei + pad

    int grp = blockIdx.x;
    if (grp >= nGroups) return;
    int lane = threadIdx.x;
    int e0 = grp * (NT * TILE_E);

    for (int i = lane; i < NT * TILE_E; i += 32) {
        int e = e0 + i; if (e >= nEdges) e = nEdges - 1;
        srcL[i] = (int)src[e];
        dstL[i] = (int)dst[e];
    }
    __syncthreads();

    // Gather cen & nei f16 rows into LDS: [row][0:128]=cen, [128:256]=nei
    const uint4* xh4 = (const uint4*)xh;   // one row = 16 x uint4 (256B)
    for (int i = lane; i < NT * TILE_E * 16; i += 32) {
        int row = i >> 4, c = i & 15;
        uint4 vc = xh4[(size_t)srcL[row] * 16 + c];
        uint4 vn = xh4[(size_t)dstL[row] * 16 + c];
        *(uint4*)&Abuf[row][c * 8]       = vc;
        *(uint4*)&Abuf[row][128 + c * 8] = vn;
    }
    __syncthreads();

    int khalf = lane >> 4;
    int nlo   = lane & 15;

#pragma unroll 1
    for (int j = 0; j < 8; ++j) {            // output column tile (N = j*16 .. j*16+15)
        // Hoist B fragments for all 8 K-steps of this column tile (reused NT times)
        v16h bfr[8];
#pragma unroll
        for (int kk = 0; kk < 8; ++kk) {
#pragma unroll
            for (int i = 0; i < 16; ++i) {
                int krow = kk * 32 + khalf * 8 + (i < 8 ? i : i + 8);  // B row K (table order)
                bfr[kk][i] = wmaskh[(size_t)krow * FDIM + j * 16 + nlo];
            }
        }
#pragma unroll
        for (int t = 0; t < NT; ++t) {
            const _Float16* myArow = &Abuf[t * TILE_E + nlo][0];  // A row m = lane&15
            v8f acc = {};
#pragma unroll
            for (int kk = 0; kk < 8; ++kk) {
                v16h a = load_A_frag(myArow, kk * 32 + khalf * 8);
                acc = __builtin_amdgcn_wmma_f32_16x16x32_f16(
                    false, a, false, bfr[kk], (short)0, acc, false, false);
            }
            // C layout: acc[r] = (M = r + 8*khalf, N = nlo) of this 16x16 tile
            int f = j * 16 + nlo;            // feature column 0..127
            int rowBase = t * TILE_E + khalf * 8;
            int curS = srcL[rowBase];
            float accum = 0.0f;
#pragma unroll
            for (int r = 0; r < 8; ++r) {
                int M = rowBase + r;
                float mval = fast_sigmoid(acc[r]);
                unsigned e = (unsigned)(e0 + M);
                mval = keep_elem(e * 128u + (unsigned)f) ? mval * 2.5f : 0.0f;
                float neif = (float)Abuf[M][128 + f];
                float v = mval * neif;
                int s = srcL[M];
                if (s != curS) {             // run-length merge (src is sorted)
                    atomicAdd(&xnew[(size_t)curS * FDIM + f], accum);
                    curS = s; accum = 0.0f;
                }
                accum += v;
            }
            atomicAdd(&xnew[(size_t)curS * FDIM + f], accum);
        }
    }
}

// ---------- K2: support = xnew @ w0 (50000x128 @ 128x128) ----------
__global__ __launch_bounds__(32, 4) void k_support(
    const float* __restrict__ xnew,
    const _Float16* __restrict__ w0h,      // [128,128] f16 row-major
    float* __restrict__ support,
    int nNodes, int nTiles) {
    __shared__ __align__(16) _Float16 Abuf[16][136];

    int tile = blockIdx.x;
    if (tile >= nTiles) return;
    int lane = threadIdx.x;
    int m0 = tile * 16;

    const float4* x4 = (const float4*)xnew;
    for (int i = lane; i < 16 * 32; i += 32) {
        int row = i >> 5, c4 = i & 31;
        int grow = m0 + row; if (grow >= nNodes) grow = nNodes - 1;
        float4 v = x4[(size_t)grow * 32 + c4];
        half4 h = { (_Float16)v.x, (_Float16)v.y, (_Float16)v.z, (_Float16)v.w };
        *(half4*)&Abuf[row][c4 * 4] = h;
    }
    __syncthreads();

    int khalf = lane >> 4;
    int nlo   = lane & 15;
    const _Float16* myArow = &Abuf[nlo][0];

#pragma unroll 1
    for (int j = 0; j < 8; ++j) {
        v8f acc = {};
#pragma unroll
        for (int kk = 0; kk < 4; ++kk) {     // K = 128 in steps of 32
            v16h a = load_A_frag(myArow, kk * 32 + khalf * 8);
            v16h b;
#pragma unroll
            for (int i = 0; i < 16; ++i) {
                int krow = kk * 32 + khalf * 8 + (i < 8 ? i : i + 8);
                b[i] = w0h[(size_t)krow * FDIM + j * 16 + nlo];
            }
            acc = __builtin_amdgcn_wmma_f32_16x16x32_f16(
                false, a, false, b, (short)0, acc, false, false);
        }
#pragma unroll
        for (int r = 0; r < 8; ++r) {
            int row = m0 + khalf * 8 + r;
            if (row < nNodes)
                support[(size_t)row * FDIM + j * 16 + nlo] = acc[r];
        }
    }
}

// ---------- K3: out[src] += adj * support[dst]  (segment_sum over sorted src) ----------
__global__ __launch_bounds__(256) void k_spmm(
    const float* __restrict__ support,
    const long long* __restrict__ src,
    const long long* __restrict__ dst,
    const float* __restrict__ adj,
    float* __restrict__ out,
    int nEdges) {
    __shared__ int   sS[128];
    __shared__ int   sD[128];
    __shared__ float sA[128];

    int t = threadIdx.x;
    int e0 = blockIdx.x * 128;
    if (t < 128) {
        int e = e0 + t;
        if (e < nEdges) { sS[t] = (int)src[e]; sD[t] = (int)dst[e]; sA[t] = adj[e]; }
        else            { sS[t] = -1; sD[t] = 0; sA[t] = 0.0f; }
    }
    __syncthreads();

    const float4* sup4 = (const float4*)support;
    int grp = t >> 5;           // 8 groups of 16 edges per block
    int col = t & 31;           // float4 column (4 features)
    int base = grp * 16;

    int curS = sS[base];
    float ax = 0.f, ay = 0.f, az = 0.f, aw = 0.f;
#pragma unroll 4
    for (int i = 0; i < 16; ++i) {
        int li = base + i;
        int s = sS[li];
        float w = sA[li];
        float4 v = sup4[(size_t)sD[li] * 32 + col];
        if (s != curS) {
            if (curS >= 0) {
                float* o = &out[(size_t)curS * FDIM + col * 4];
                atomicAdd(o + 0, ax); atomicAdd(o + 1, ay);
                atomicAdd(o + 2, az); atomicAdd(o + 3, aw);
            }
            curS = s; ax = ay = az = aw = 0.f;
        }
        ax += v.x * w; ay += v.y * w; az += v.z * w; aw += v.w * w;
    }
    if (curS >= 0) {
        float* o = &out[(size_t)curS * FDIM + col * 4];
        atomicAdd(o + 0, ax); atomicAdd(o + 1, ay);
        atomicAdd(o + 2, az); atomicAdd(o + 3, aw);
    }
}

// ---------- launcher ----------
extern "C" void kernel_launch(void* const* d_in, const int* in_sizes, int n_in,
                              void* d_out, int out_size, void* d_ws, size_t ws_size,
                              hipStream_t stream) {
    const float*     x     = (const float*)d_in[0];
    const long long* src   = (const long long*)d_in[1];   // int64 per reference
    const long long* dst   = (const long long*)d_in[2];
    const float*     adj   = (const float*)d_in[3];
    const float*     wmask = (const float*)d_in[4];
    const float*     w0    = (const float*)d_in[5];
    float*           out   = (float*)d_out;

    int nNodeF = in_sizes[0];          // N * 128
    int nNodes = nNodeF / FDIM;
    int nEdges = in_sizes[1];

    // workspace layout (256B aligned)
    char*  ws   = (char*)d_ws;
    size_t offA = 0;                                                 // xnew    f32 [N,128]
    size_t offB = (offA + (size_t)nNodeF * 4 + 255) & ~(size_t)255;  // support f32 [N,128]
    size_t offC = (offB + (size_t)nNodeF * 4 + 255) & ~(size_t)255;  // xh      f16 [N,128]
    size_t offD = (offC + (size_t)nNodeF * 2 + 255) & ~(size_t)255;  // wmask   f16 [256,128]
    size_t offE = (offD + (size_t)2 * FDIM * FDIM * 2 + 255) & ~(size_t)255; // w0 f16
    float*     xnew    = (float*)(ws + offA);
    float*     support = (float*)(ws + offB);
    _Float16*  xh      = (_Float16*)(ws + offC);
    _Float16*  wmaskh  = (_Float16*)(ws + offD);
    _Float16*  w0h     = (_Float16*)(ws + offE);

    k_init<<<(nNodeF + 255) / 256, 256, 0, stream>>>(
        x, wmask, w0, xnew, xh, wmaskh, w0h, out, nNodeF);

    int nGroups = (nEdges + NT * TILE_E - 1) / (NT * TILE_E);
    k_mask_agg<<<nGroups, 32, 0, stream>>>(xh, src, dst, wmaskh, xnew, nEdges, nGroups);

    int nTilesM = (nNodes + 15) / 16;
    k_support<<<nTilesM, 32, 0, stream>>>(xnew, w0h, support, nNodes, nTilesM);

    int nBlocksS = (nEdges + 127) / 128;
    k_spmm<<<nBlocksS, 256, 0, stream>>>(support, src, dst, adj, out, nEdges);
}
